// RNNLayer_73744588472745
// MI455X (gfx1250) — matrix operand order repaired and checked by reading, not compiled, not run on hardware
//
#include <hip/hip_runtime.h>

// ---------------------------------------------------------------------------
// Vanilla RNN: out[b,t,:] = tanh(x[b,t,:] @ Wx + b + h_{t-1} @ Wh)
// B=64, T=1024, D=1024, H=1024, f32 in/out; bf16 WMMA (f32 accum).
// Phase 1: x_proj GEMM into d_out (register-double-buffered staging).
// Phase 2: sequential scan in-place; bf16 mirror of h_t in d_ws async-DMA'd
//          into LDS (GLOBAL_LOAD_ASYNC_TO_LDS_B128) each timestep.
// ---------------------------------------------------------------------------

typedef __attribute__((ext_vector_type(16))) __bf16 v16bf;
typedef __attribute__((ext_vector_type(8)))  __bf16 v8bf;
typedef __attribute__((ext_vector_type(8)))  float  v8f;
typedef __attribute__((ext_vector_type(4)))  int    v4i;

#define B_  64
#define T_  1024
#define D_  1024
#define H_  1024

#define AS3 __attribute__((address_space(3)))
#define AS1 __attribute__((address_space(1)))

#if __has_builtin(__builtin_amdgcn_global_load_async_to_lds_b128)
#define HAVE_ASYNC_LDS 1
#else
#define HAVE_ASYNC_LDS 0
#endif

__device__ __forceinline__ void wait_async0() {
#if __has_builtin(__builtin_amdgcn_s_wait_asynccnt)
    __builtin_amdgcn_s_wait_asynccnt(0);
#else
    asm volatile("s_wait_asynccnt 0" ::: "memory");
#endif
}

__device__ __forceinline__ __bf16 f2bf(float f) {
    unsigned u = __builtin_bit_cast(unsigned, f);
    unsigned short s = (unsigned short)(u >> 16);   // truncate-to-bf16
    return __builtin_bit_cast(__bf16, s);
}

// pack two floats into a dword holding {k, k+1} bf16 pair (little-endian)
__device__ __forceinline__ unsigned pack2(float lo, float hi) {
    unsigned a = __builtin_bit_cast(unsigned, lo) >> 16;
    unsigned b = __builtin_bit_cast(unsigned, hi) & 0xffff0000u;
    return a | b;
}

__device__ __forceinline__ v8bf pack8(float4 a, float4 b) {
    v8bf r;
    r[0] = f2bf(a.x); r[1] = f2bf(a.y); r[2] = f2bf(a.z); r[3] = f2bf(a.w);
    r[4] = f2bf(b.x); r[5] = f2bf(b.y); r[6] = f2bf(b.z); r[7] = f2bf(b.w);
    return r;
}

__device__ __forceinline__ v16bf cat16(v8bf a, v8bf b) {
    return __builtin_shufflevector(a, b, 0,1,2,3,4,5,6,7,8,9,10,11,12,13,14,15);
}

// ---------------------------------------------------------------------------
// Phase 1: x_proj = x @ Wx + bias  (M=65536, K=1024, N=1024) -> out
// Block 256 thr (8 waves); tile M=64 x N=128, K-step 32.
// Global->register prefetch of the next K tile overlaps the WMMAs.
// ---------------------------------------------------------------------------
__global__ __launch_bounds__(256)
void xproj_gemm(const float* __restrict__ x, const float* __restrict__ W,
                const float* __restrict__ bias, float* __restrict__ out)
{
    __shared__ alignas(16) __bf16 Ald [64][40];    // A tile, row-major (M x K)
    __shared__ alignas(16) __bf16 BldT[128][40];   // B tile, K-major (N x K)

    const int tid  = threadIdx.x;
    const int lane = tid & 31;
    const int wave = tid >> 5;
    const int lr   = lane & 15;
    const int hi   = lane >> 4;

    const int mblk = blockIdx.x * 64;
    const int nblk = blockIdx.y * 128;
    const int msub = (wave & 3) * 16;
    const int nhal = (wave >> 2) * 64;

    // this thread's staging slice
    const int ar  = tid >> 2;            // A row 0..63
    const int ac  = (tid & 3) * 8;       // A col base 0..24
    const int br0 = (tid >> 4) * 2;      // B row pair 0,2,..,30
    const int bc0 = (tid & 15) * 8;      // B col base 0..120
    const float* arow = &x[(size_t)(mblk + ar) * D_];      // + k0 + ac
    const float* bcol = &W[nblk + bc0];                    // + (k0+br0)*H_

    v8f acc[4] = {{}, {}, {}, {}};

    // prologue: load k0 = 0 tile into registers
    float4 a0r = ((const float4*)(arow + ac))[0];
    float4 a1r = ((const float4*)(arow + ac))[1];
    float4 p0  = ((const float4*)(bcol + (size_t)br0 * H_))[0];
    float4 p1  = ((const float4*)(bcol + (size_t)br0 * H_))[1];
    float4 q0  = ((const float4*)(bcol + (size_t)(br0 + 1) * H_))[0];
    float4 q1  = ((const float4*)(bcol + (size_t)(br0 + 1) * H_))[1];

    for (int k0 = 0; k0 < D_; k0 += 32) {
        // --- stage current registers into LDS
        *(v8bf*)&Ald[ar][ac] = pack8(a0r, a1r);
        {
            float r0v[8] = {p0.x,p0.y,p0.z,p0.w,p1.x,p1.y,p1.z,p1.w};
            float r1v[8] = {q0.x,q0.y,q0.z,q0.w,q1.x,q1.y,q1.z,q1.w};
            #pragma unroll
            for (int j = 0; j < 8; ++j)
                *(unsigned*)&BldT[bc0 + j][br0] = pack2(r0v[j], r1v[j]);
        }
        __syncthreads();

        // --- prefetch next K tile into registers (overlaps the WMMAs)
        if (k0 + 32 < D_) {
            const int kn = k0 + 32;
            a0r = ((const float4*)(arow + kn + ac))[0];
            a1r = ((const float4*)(arow + kn + ac))[1];
            p0  = ((const float4*)(bcol + (size_t)(kn + br0) * H_))[0];
            p1  = ((const float4*)(bcol + (size_t)(kn + br0) * H_))[1];
            q0  = ((const float4*)(bcol + (size_t)(kn + br0 + 1) * H_))[0];
            q1  = ((const float4*)(bcol + (size_t)(kn + br0 + 1) * H_))[1];
        }

        // --- fragments: A once, all four B frags batched, then 4 WMMAs
        v8bf a0 = *(const v8bf*)&Ald[msub + lr][hi * 16];
        v8bf a1 = *(const v8bf*)&Ald[msub + lr][hi * 16 + 8];
        v16bf a = cat16(a0, a1);

        v16bf bfr[4];
        #pragma unroll
        for (int nt = 0; nt < 4; ++nt) {
            const __bf16* bp = &BldT[nhal + nt * 16 + lr][hi * 16];
            bfr[nt] = cat16(((const v8bf*)bp)[0], ((const v8bf*)bp)[1]);
        }
        #pragma unroll
        for (int nt = 0; nt < 4; ++nt)
            acc[nt] = __builtin_amdgcn_wmma_f32_16x16x32_bf16(false, a, false, bfr[nt],
                                                              (short)0, acc[nt],
                                                              false, false);
        __syncthreads();
    }

    #pragma unroll
    for (int nt = 0; nt < 4; ++nt) {
        #pragma unroll
        for (int r = 0; r < 8; ++r) {
            int row = mblk + msub + hi * 8 + r;
            int col = nblk + nhal + nt * 16 + lr;
            out[(size_t)row * H_ + col] = acc[nt][r] + bias[col];
        }
    }
}

// ---------------------------------------------------------------------------
// Phase 2: sequential scan. Grid = 4 batch-groups x 32 col-groups.
// Block 256 thr (8 waves) = 2 N-tiles x 4 K-quarters.
// Wh slice kept K-major in LDS (bf16) for the whole scan.
// h_t mirrored as bf16 in hbuf (d_ws) -> async DMA into LDS next step.
// ---------------------------------------------------------------------------
__global__ __launch_bounds__(256)
void rnn_scan(const float* __restrict__ W, float* __restrict__ out,
              unsigned* __restrict__ cnt, __bf16* __restrict__ hbuf)
{
    __shared__ alignas(16) __bf16 WhlT [32][1032];   // Wh[:, n0+c] K-major ~66KB
    __shared__ alignas(16) __bf16 hprev[16][1032];   // h_{t-1} rows        ~33KB
    __shared__ alignas(16) float  partial[16][34];   // K-split reduction

    const int tid  = threadIdx.x;
    const int lane = tid & 31;
    const int wave = tid >> 5;
    const int lr   = lane & 15;
    const int hi   = lane >> 4;

    const int bg = blockIdx.x;          // 0..3  (16 batch rows)
    const int cg = blockIdx.y;          // 0..31 (32 H columns)
    const int n0 = cg * 32;

    const int ntile = wave & 1;         // 16-col tile
    const int kq    = wave >> 1;        // 256-deep K quarter

    // preload Wh slice once, transposed to K-major, paired-b32 stores
    for (int j = 0; j < 64; ++j) {
        int pidx = tid + j * 256;       // row-pair x col
        int c    = pidx & 31;
        int r0   = (pidx >> 5) * 2;     // 0,2,..,1022
        float v0 = W[(size_t)(D_ + r0    ) * H_ + n0 + c];
        float v1 = W[(size_t)(D_ + r0 + 1) * H_ + n0 + c];
        *(unsigned*)&WhlT[c][r0] = pack2(v0, v1);
    }
    __syncthreads();

    const int hr = tid >> 4;            // 0..15  hprev staging row
    const int hc = (tid & 15) * 64;     // 0..960 hprev staging col base

    for (int t = 0; t < T_; ++t) {
        // stage h_{t-1} (16 x 1024 bf16) into LDS
        if (t == 0) {
            v8bf z = {};
            #pragma unroll
            for (int u = 0; u < 8; ++u)
                *(v8bf*)&hprev[hr][hc + u * 8] = z;
        } else {
#if HAVE_ASYNC_LDS
            const __bf16* src = hbuf + (size_t)(bg * 16 + hr) * H_ + hc;
            #pragma unroll
            for (int u = 0; u < 8; ++u) {
                __builtin_amdgcn_global_load_async_to_lds_b128(
                    (AS1 v4i*)(src + u * 8),
                    (AS3 v4i*)&hprev[hr][hc + u * 8],
                    /*offset=*/0, /*cpol=*/0);
            }
            wait_async0();
#else
            const float4* src = (const float4*)
                &out[((size_t)(bg * 16 + hr) * T_ + (t - 1)) * H_ + hc];
            #pragma unroll
            for (int u = 0; u < 8; ++u)
                *(v8bf*)&hprev[hr][hc + u * 8] = pack8(src[2*u], src[2*u + 1]);
#endif
        }
        for (int idx = tid; idx < 16 * 34; idx += 256)
            (&partial[0][0])[idx] = 0.0f;
        __syncthreads();

        // 16x32 = hprev(16x1024) @ Wh(1024x32), this wave's K quarter
        v8f acc = {};
        #pragma unroll
        for (int ks = 0; ks < 8; ++ks) {
            const int kb = kq * 256 + ks * 32;
            v8bf a0 = *(const v8bf*)&hprev[lr][kb + hi * 16];
            v8bf a1 = *(const v8bf*)&hprev[lr][kb + hi * 16 + 8];
            v16bf a = cat16(a0, a1);
            const __bf16* bp = &WhlT[ntile * 16 + lr][kb + hi * 16];
            v16bf b = cat16(((const v8bf*)bp)[0], ((const v8bf*)bp)[1]);
            acc = __builtin_amdgcn_wmma_f32_16x16x32_bf16(false, a, false, b,
                                                          (short)0, acc,
                                                          false, false);
        }

        // reduce K quarters with LDS float atomics (ds_add_f32)
        #pragma unroll
        for (int r = 0; r < 8; ++r)
            atomicAdd(&partial[hi * 8 + r][ntile * 16 + lr], acc[r]);
        __syncthreads();

        // all 8 waves finalize 2 elems each: h = tanh(xp + sum)
        {
            int e   = tid * 2;                    // pairs within a row
            int row = e >> 5;                     // 0..15 local batch row
            int lc  = e & 31;                     // even local col
            int bat = bg * 16 + row;
            size_t o = ((size_t)bat * T_ + t) * H_ + (n0 + lc);
            float2 xp = *(const float2*)&out[o];
            float h0 = tanhf(xp.x + partial[row][lc]);
            float h1 = tanhf(xp.y + partial[row][lc + 1]);
            float2 hv; hv.x = h0; hv.y = h1;
            *(float2*)&out[o] = hv;                                   // f32 output
            *(unsigned*)&hbuf[(size_t)bat * H_ + n0 + lc] = pack2(h0, h1); // bf16 mirror
        }
        __syncthreads();

        // per-batch-group grid barrier (32 blocks), monotonic counter
        if (tid == 0) {
            __threadfence();
            __hip_atomic_fetch_add(&cnt[bg], 1u, __ATOMIC_ACQ_REL,
                                   __HIP_MEMORY_SCOPE_AGENT);
            const unsigned target = 32u * (unsigned)(t + 1);
            while (__hip_atomic_load(&cnt[bg], __ATOMIC_ACQUIRE,
                                     __HIP_MEMORY_SCOPE_AGENT) < target) {
                __builtin_amdgcn_s_sleep(2);
            }
        }
        __syncthreads();
    }
}

__global__ void init_cnt(unsigned* c) {
    if (threadIdx.x < 4) c[threadIdx.x] = 0u;
}

// ---------------------------------------------------------------------------
extern "C" void kernel_launch(void* const* d_in, const int* in_sizes, int n_in,
                              void* d_out, int out_size, void* d_ws, size_t ws_size,
                              hipStream_t stream)
{
    const float* x    = (const float*)d_in[0];   // [B,T,D]
    const float* W    = (const float*)d_in[1];   // [D+H, H]
    const float* bias = (const float*)d_in[2];   // [H]
    float*       out  = (float*)d_out;           // [B,T,H]
    unsigned*    cnt  = (unsigned*)d_ws;                       // 16 B
    __bf16*      hbuf = (__bf16*)((char*)d_ws + 256);          // 64x1024 bf16

    init_cnt<<<1, 32, 0, stream>>>(cnt);

    dim3 g1((B_ * T_) / 64, H_ / 128);           // 1024 x 8 blocks
    xproj_gemm<<<g1, 256, 0, stream>>>(x, W, bias, out);

    dim3 g2(4, 32);                              // batch-groups x col-groups
    rnn_scan<<<g2, 256, 0, stream>>>(W, out, cnt, hbuf);
}
